// RatioOfDistancesMetric_39170101739936
// MI455X (gfx1250) — matrix operand order repaired and checked by reading, not compiled, not run on hardware
//
#include <hip/hip_runtime.h>
#include <hip/hip_bf16.h>
#include <stdint.h>

// ---------------------------------------------------------------------------
// Types
// ---------------------------------------------------------------------------
typedef __bf16 bf16_t;
typedef __bf16 v8bf  __attribute__((ext_vector_type(8)));
typedef __bf16 v16bf __attribute__((ext_vector_type(16)));
typedef float  v8f   __attribute__((ext_vector_type(8)));
typedef int    v4i   __attribute__((ext_vector_type(4)));

#define N_SAMPLES 4096
#define N_TRAIN   65536
#define DIMK      256
#define BLK_M     64
#define BLK_N     64
#define NSPLIT    8
#define NSTEPS    ((N_TRAIN / NSPLIT) / BLK_N)   // 128 tile steps per block
#define LDK       (DIMK + 8)                     // +16B pad: kills LDS bank conflicts

// ---------------------------------------------------------------------------
// Workspace layout (bytes)
// ---------------------------------------------------------------------------
static constexpr size_t SZ_THI  = (size_t)N_TRAIN * DIMK * 2;   // 32 MB
static constexpr size_t SZ_SHI  = (size_t)N_SAMPLES * DIMK * 2; // 2 MB
static constexpr size_t OFF_THI = 0;
static constexpr size_t OFF_TLO = OFF_THI + SZ_THI;
static constexpr size_t OFF_SHI = OFF_TLO + SZ_THI;
static constexpr size_t OFF_SLO = OFF_SHI + SZ_SHI;
static constexpr size_t OFF_XT2 = OFF_SLO + SZ_SHI;
static constexpr size_t OFF_XS2 = OFF_XT2 + (size_t)N_TRAIN * 4;
static constexpr size_t OFF_CNT = OFF_XS2 + (size_t)N_SAMPLES * 4;
static constexpr size_t OFF_PART = OFF_CNT + 256;               // aligned
// part: [N_SAMPLES][NSPLIT][2] floats

// ---------------------------------------------------------------------------
// CDNA5 async global->LDS copy (guarded; falls back to sync copy)
// ---------------------------------------------------------------------------
#if defined(__gfx1250__) && __has_builtin(__builtin_amdgcn_global_load_async_to_lds_b128)
#define HAVE_ASYNC 1
#else
#define HAVE_ASYNC 0
#endif

#if HAVE_ASYNC
typedef __attribute__((address_space(1))) v4i gbl_v4i;  // "__device__" AS
typedef __attribute__((address_space(3))) v4i lds_v4i;  // "__shared__" AS
__device__ __forceinline__ void async_copy16(const bf16_t* g, bf16_t* l) {
  __builtin_amdgcn_global_load_async_to_lds_b128((gbl_v4i*)g, (lds_v4i*)l, 0, 0);
}
#endif

__device__ __forceinline__ void wait_async_copies() {
#if HAVE_ASYNC
#if __has_builtin(__builtin_amdgcn_s_wait_asynccnt)
  __builtin_amdgcn_s_wait_asynccnt(0);
#else
  asm volatile("s_wait_asynccnt 0" ::: "memory");
#endif
#endif
}

// ---------------------------------------------------------------------------
// WMMA helper: D = A*B + C  (bf16 in, f32 acc)
// ---------------------------------------------------------------------------
#define WMMA_BF16(A, B, C) \
  __builtin_amdgcn_wmma_f32_16x16x32_bf16(false, (A), false, (B), (short)0, (C), false, false)

// A-fragment (16-bit A 16x32): per lane, K = kb..kb+7 and kb+16..kb+23 (kb=(lane/16)*8)
__device__ __forceinline__ v16bf ld_frag_a(const bf16_t* p) {
  v8bf x = *(const v8bf*)p;
  v8bf y = *(const v8bf*)(p + 16);
  v16bf r;
#pragma unroll
  for (int i = 0; i < 8; ++i) { r[i] = x[i]; r[i + 8] = y[i]; }
  return r;
}
// B-fragment (16-bit B 32x16): per lane, K = kb..kb+7 and kb+16..kb+23 (kb=(lane/16)*8)
__device__ __forceinline__ v16bf ld_frag_b(const bf16_t* p) {
  v8bf x = *(const v8bf*)p;
  v8bf y = *(const v8bf*)(p + 16);
  v16bf r;
#pragma unroll
  for (int i = 0; i < 8; ++i) { r[i] = x[i]; r[i + 8] = y[i]; }
  return r;
}

// sorted-pair top2 merge: (m1<=m2) merged with (b1<=b2)
__device__ __forceinline__ void top2_merge(float& m1, float& m2, float b1, float b2) {
  float lo = fminf(m1, b1);
  float hi = fmaxf(m1, b1);
  m2 = fminf(hi, fminf(m2, b2));
  m1 = lo;
}
// push one scalar candidate into sorted pair (3 VALU ops)
__device__ __forceinline__ void top2_push(float& m1, float& m2, float v) {
  float hi = fmaxf(m1, v);
  m1 = fminf(m1, v);
  m2 = fminf(m2, hi);
}

// ---------------------------------------------------------------------------
// Kernel 1: fp32 -> bf16 hi/lo split + squared row norms; zero the counter
// One 256-thread block per row (train rows first, then sample rows)
// ---------------------------------------------------------------------------
__global__ __launch_bounds__(256) void prep_kernel(
    const float* __restrict__ Xs, const float* __restrict__ Xt,
    bf16_t* __restrict__ tHi, bf16_t* __restrict__ tLo,
    bf16_t* __restrict__ sHi, bf16_t* __restrict__ sLo,
    float* __restrict__ xt2, float* __restrict__ xs2, int* __restrict__ cnt) {
  const int row = blockIdx.x;
  const int tid = threadIdx.x;
  const float* src; bf16_t* dh; bf16_t* dl; float* nrm; int r;
  if (row < N_TRAIN) { src = Xt; dh = tHi; dl = tLo; nrm = xt2; r = row; }
  else               { src = Xs; dh = sHi; dl = sLo; nrm = xs2; r = row - N_TRAIN; }

  const size_t idx = (size_t)r * DIMK + tid;
  const float x = src[idx];
  const bf16_t h = (bf16_t)x;
  dh[idx] = h;
  dl[idx] = (bf16_t)(x - (float)h);

  __shared__ float red[256];
  red[tid] = x * x;
  __syncthreads();
#pragma unroll
  for (int s = 128; s > 0; s >>= 1) {
    if (tid < s) red[tid] += red[tid + s];
    __syncthreads();
  }
  if (tid == 0) nrm[r] = red[0];
  if (row == 0 && tid == 0) *cnt = 0;
}

// ---------------------------------------------------------------------------
// Kernel 2: fused split-bf16 WMMA GEMM + per-row top-2 of (xt2[n] - 2*dot)
// Grid: (N_SAMPLES/BLK_M, NSPLIT), 256 threads = 8 waves (2 M-waves x 4 N-waves)
// Running top-2 kept in registers per lane; single cross-lane butterfly at end.
// ---------------------------------------------------------------------------
__global__ __launch_bounds__(256) void gemm_top2_kernel(
    const bf16_t* __restrict__ tHi, const bf16_t* __restrict__ tLo,
    const bf16_t* __restrict__ sHi, const bf16_t* __restrict__ sLo,
    const float* __restrict__ xt2, float* __restrict__ part) {
  __shared__ bf16_t Ah[BLK_M][LDK];
  __shared__ bf16_t Al[BLK_M][LDK];
  __shared__ bf16_t Bh[BLK_N][LDK];
  __shared__ bf16_t Bl[BLK_N][LDK];
  __shared__ float  sTop2[4][BLK_M][2];

  const int tid  = threadIdx.x;
  const int lane = tid & 31;
  const int wid  = tid >> 5;
  const int wm   = wid & 1;        // M wave row (0..1)
  const int wn   = wid >> 1;       // N wave col (0..3)
  const int lr   = lane & 15;
  const int lh   = lane >> 4;

  const int sbase    = blockIdx.x * BLK_M;
  const int nsplbase = blockIdx.y * (N_TRAIN / NSPLIT);

  // Stage sample block (hi/lo) once: 64 rows x 512B, 16B chunks
  for (int i = tid; i < BLK_M * 32; i += 256) {
    const int row = i >> 5, c = (i & 31) * 8;
    const size_t g = (size_t)(sbase + row) * DIMK + c;
#if HAVE_ASYNC
    async_copy16(sHi + g, &Ah[row][c]);
    async_copy16(sLo + g, &Al[row][c]);
#else
    *(v8bf*)&Ah[row][c] = *(const v8bf*)(sHi + g);
    *(v8bf*)&Al[row][c] = *(const v8bf*)(sLo + g);
#endif
  }
  wait_async_copies();
  __syncthreads();

  const int am0  = wm * 32 + lr;        // A tile 0 row for this lane
  const int am1  = am0 + 16;            // A tile 1 row
  const int brow = wn * 16 + lr;        // B column (= train row within tile)

  // Per-lane running top-2 of (xt2[n] - 2*dot) for each of this lane's C rows
  float run1[2][8], run2[2][8];
#pragma unroll
  for (int t = 0; t < 2; ++t)
#pragma unroll
    for (int r = 0; r < 8; ++r) { run1[t][r] = __builtin_inff(); run2[t][r] = __builtin_inff(); }

  for (int nb = 0; nb < NSTEPS; ++nb) {
    const int nbase = nsplbase + nb * BLK_N;

    // Stage train tile (hi/lo)
    for (int i = tid; i < BLK_N * 32; i += 256) {
      const int row = i >> 5, c = (i & 31) * 8;
      const size_t g = (size_t)(nbase + row) * DIMK + c;
#if HAVE_ASYNC
      async_copy16(tHi + g, &Bh[row][c]);
      async_copy16(tLo + g, &Bl[row][c]);
#else
      *(v8bf*)&Bh[row][c] = *(const v8bf*)(tHi + g);
      *(v8bf*)&Bl[row][c] = *(const v8bf*)(tLo + g);
#endif
    }
    wait_async_copies();
    __syncthreads();

    v8f acc0 = {};
    v8f acc1 = {};
    for (int ks = 0; ks < 8; ++ks) {
      const int kc = ks * 32;
      v16bf a0h = ld_frag_a(&Ah[am0][kc + lh * 8]);
      v16bf a0l = ld_frag_a(&Al[am0][kc + lh * 8]);
      v16bf a1h = ld_frag_a(&Ah[am1][kc + lh * 8]);
      v16bf a1l = ld_frag_a(&Al[am1][kc + lh * 8]);
      v16bf bh  = ld_frag_b(&Bh[brow][kc + lh * 8]);
      v16bf bl  = ld_frag_b(&Bl[brow][kc + lh * 8]);
      // split-precision: hi*hi + hi*lo + lo*hi
      acc0 = WMMA_BF16(a0h, bh, acc0);
      acc0 = WMMA_BF16(a0h, bl, acc0);
      acc0 = WMMA_BF16(a0l, bh, acc0);
      acc1 = WMMA_BF16(a1h, bh, acc1);
      acc1 = WMMA_BF16(a1h, bl, acc1);
      acc1 = WMMA_BF16(a1l, bh, acc1);
    }

    // fold this tile's C values into the per-lane running top-2 (3 ops each)
    const float xt2v = xt2[nbase + brow];
#pragma unroll
    for (int t = 0; t < 2; ++t) {
      v8f& a = t ? acc1 : acc0;
#pragma unroll
      for (int r = 0; r < 8; ++r)
        top2_push(run1[t][r], run2[t][r], fmaf(-2.0f, a[r], xt2v));
    }
    __syncthreads();  // all waves done reading Bh/Bl before next stage
  }

  // One cross-lane butterfly per C row: top-2 over the 16 N-lanes
#pragma unroll
  for (int t = 0; t < 2; ++t) {
#pragma unroll
    for (int r = 0; r < 8; ++r) {
      float m1 = run1[t][r], m2 = run2[t][r];
#pragma unroll
      for (int mask = 1; mask <= 8; mask <<= 1) {
        const float o1 = __shfl_xor(m1, mask, 32);
        const float o2 = __shfl_xor(m2, mask, 32);
        top2_merge(m1, m2, o1, o2);
      }
      if (lr == 0) {  // lanes 0 (rows m=r) and 16 (rows m=8+r) hold results
        const int mrow = wm * 32 + t * 16 + lh * 8 + r;
        sTop2[wn][mrow][0] = m1;   // unique writer per slot: plain store
        sTop2[wn][mrow][1] = m2;
      }
    }
  }
  __syncthreads();

  // Merge the 4 N-wave columns, spill block-level top2 partials
  if (tid < BLK_M) {
    float m1 = __builtin_inff(), m2 = __builtin_inff();
#pragma unroll
    for (int c = 0; c < 4; ++c)
      top2_merge(m1, m2, sTop2[c][tid][0], sTop2[c][tid][1]);
    float* p = part + ((size_t)(sbase + tid) * NSPLIT + blockIdx.y) * 2;
    p[0] = m1;
    p[1] = m2;
  }
}

// ---------------------------------------------------------------------------
// Kernel 3: merge N-splits, finish distances, count memorized
// ---------------------------------------------------------------------------
__global__ __launch_bounds__(256) void merge_kernel(
    const float* __restrict__ part, const float* __restrict__ xs2,
    float* __restrict__ out, int* __restrict__ cnt) {
  const int s = blockIdx.x * blockDim.x + threadIdx.x;
  if (s >= N_SAMPLES) return;
  float m1 = __builtin_inff(), m2 = __builtin_inff();
#pragma unroll
  for (int i = 0; i < NSPLIT; ++i) {
    const float* p = part + ((size_t)s * NSPLIT + i) * 2;
    top2_merge(m1, m2, p[0], p[1]);
  }
  const float x2 = xs2[s];
  const float d0 = sqrtf(fmaxf(x2 + m1, 0.0f));
  const float d1 = sqrtf(fmaxf(x2 + m2, 0.0f));
  out[2 * s]     = d0;
  out[2 * s + 1] = d1;
  if (d0 < (1.0f / 3.0f) * d1) atomicAdd(cnt, 1);
}

__global__ void ratio_kernel(const int* __restrict__ cnt, float* __restrict__ out) {
  out[N_SAMPLES * 2] = (float)(*cnt) * (1.0f / (float)N_SAMPLES);
}

// ---------------------------------------------------------------------------
// Launch
// ---------------------------------------------------------------------------
extern "C" void kernel_launch(void* const* d_in, const int* in_sizes, int n_in,
                              void* d_out, int out_size, void* d_ws, size_t ws_size,
                              hipStream_t stream) {
  const float* Xs = (const float*)d_in[0];  // [4096, 256]
  const float* Xt = (const float*)d_in[1];  // [65536, 256]
  float* out = (float*)d_out;               // [4096*2] distances + [1] ratio
  char* ws = (char*)d_ws;

  bf16_t* tHi = (bf16_t*)(ws + OFF_THI);
  bf16_t* tLo = (bf16_t*)(ws + OFF_TLO);
  bf16_t* sHi = (bf16_t*)(ws + OFF_SHI);
  bf16_t* sLo = (bf16_t*)(ws + OFF_SLO);
  float*  xt2 = (float*)(ws + OFF_XT2);
  float*  xs2 = (float*)(ws + OFF_XS2);
  int*    cnt = (int*)(ws + OFF_CNT);
  float*  part = (float*)(ws + OFF_PART);

  prep_kernel<<<N_TRAIN + N_SAMPLES, 256, 0, stream>>>(Xs, Xt, tHi, tLo, sHi, sLo,
                                                       xt2, xs2, cnt);
  dim3 grid(N_SAMPLES / BLK_M, NSPLIT);
  gemm_top2_kernel<<<grid, 256, 0, stream>>>(tHi, tLo, sHi, sLo, xt2, part);
  merge_kernel<<<(N_SAMPLES + 255) / 256, 256, 0, stream>>>(part, xs2, out, cnt);
  ratio_kernel<<<1, 1, 0, stream>>>(cnt, out);
}
